// PointNetFeatureExtractorFlexible_86268713107603
// MI455X (gfx1250) — compile-verified, matrix operand order checked by reading, not run on hardware
//
#include <hip/hip_runtime.h>
#include <hip/hip_bf16.h>

typedef __attribute__((ext_vector_type(16))) _Float16 v16h;
typedef __attribute__((ext_vector_type(8)))  float    v8f;
typedef __attribute__((ext_vector_type(4)))  unsigned v4u;
typedef __attribute__((ext_vector_type(8)))  int      v8i;
typedef __attribute__((ext_vector_type(4)))  int      v4i;

#define WMMA_F16(a, b, c) \
  __builtin_amdgcn_wmma_f32_16x16x32_f16(false, (a), false, (b), (short)0, (c), false, false)

#if defined(__HIP_DEVICE_COMPILE__) && __has_builtin(__builtin_amdgcn_tensor_load_to_lds)
#define HAVE_TDM 1
#else
#define HAVE_TDM 0
#endif

#if HAVE_TDM
// Issue one TDM 2D tile load: rows x tileK f16 elements, row stride strideElems,
// from gptr into LDS offset ldsOff. D# layout per CDNA5 ISA ch.8.
__device__ inline void tdm_load_2d(unsigned ldsOff, const void* gptr,
                                   int tileK, int rows, int strideElems) {
  unsigned long long ga = (unsigned long long)(uintptr_t)gptr;
  v4u g0;
  g0.x = 1u;                                           // count=1 (valid descriptor)
  g0.y = ldsOff;                                       // lds_addr
  g0.z = (unsigned)(ga & 0xFFFFFFFFu);                 // global_addr[31:0]
  g0.w = (unsigned)((ga >> 32) & 0x1FFFFFFu) | (2u << 30); // global_addr[56:32] | type=2
  v8i g1;
  g1[0] = (int)(1u << 16);                             // data_size=1 -> 2 bytes
  g1[1] = (int)(((unsigned)tileK & 0xFFFFu) << 16);    // tensor_dim0[15:0]
  g1[2] = (int)(((unsigned)rows & 0xFFFFu) << 16);     // dim0 hi=0 | tensor_dim1[15:0]
  g1[3] = (int)(((unsigned)tileK & 0xFFFFu) << 16);    // dim1 hi=0 | tile_dim0
  g1[4] = rows;                                        // tile_dim1 | tile_dim2=0
  g1[5] = strideElems;                                 // tensor_dim0_stride[31:0]
  g1[6] = 0;                                           // stride0 hi | stride1 lo
  g1[7] = 0;
  v4i z4 = {0, 0, 0, 0};
#if __clang_major__ >= 23
  v8i z8 = {0, 0, 0, 0, 0, 0, 0, 0};
  __builtin_amdgcn_tensor_load_to_lds(g0, g1, z4, z4, z8, 0);
#else
  __builtin_amdgcn_tensor_load_to_lds(g0, g1, z4, z4, 0);
#endif
}
#endif

// ---------------------------------------------------------------------------
// Farthest point sampling: one block per batch, sequential argmax selection.
// ---------------------------------------------------------------------------
__global__ void fps_kernel(const float* __restrict__ pos, int N, int M,
                           int* __restrict__ idx, float* __restrict__ centers) {
  int b = blockIdx.x;
  const float* P = pos + (size_t)b * N * 3;
  __shared__ float dmin[2048];
  __shared__ float rv[256];
  __shared__ int   ri[256];
  __shared__ float sel[3];
  int t = threadIdx.x;

  float p0x = P[0], p0y = P[1], p0z = P[2];
  for (int j = t; j < N; j += 256) {
    float dx = P[j * 3 + 0] - p0x, dy = P[j * 3 + 1] - p0y, dz = P[j * 3 + 2] - p0z;
    dmin[j] = dx * dx + dy * dy + dz * dz;
  }
  if (t == 0) {
    idx[b * M] = 0;
    centers[(size_t)(b * M) * 3 + 0] = p0x;
    centers[(size_t)(b * M) * 3 + 1] = p0y;
    centers[(size_t)(b * M) * 3 + 2] = p0z;
  }
  __syncthreads();

  for (int i = 1; i < M; ++i) {
    float bv = -1.0f; int bi = N;
    for (int j = t; j < N; j += 256) {
      float v = dmin[j];
      if (v > bv) { bv = v; bi = j; }
    }
    rv[t] = bv; ri[t] = bi;
    __syncthreads();
    for (int s = 128; s > 0; s >>= 1) {
      if (t < s) {
        float ov = rv[t + s]; int oi = ri[t + s];
        if (ov > rv[t] || (ov == rv[t] && oi < ri[t])) { rv[t] = ov; ri[t] = oi; }
      }
      __syncthreads();
    }
    if (t == 0) {
      int w = ri[0];
      idx[b * M + i] = w;
      sel[0] = P[w * 3 + 0]; sel[1] = P[w * 3 + 1]; sel[2] = P[w * 3 + 2];
      centers[(size_t)(b * M + i) * 3 + 0] = sel[0];
      centers[(size_t)(b * M + i) * 3 + 1] = sel[1];
      centers[(size_t)(b * M + i) * 3 + 2] = sel[2];
    }
    __syncthreads();
    float sx = sel[0], sy = sel[1], sz = sel[2];
    for (int j = t; j < N; j += 256) {
      float dx = P[j * 3 + 0] - sx, dy = P[j * 3 + 1] - sy, dz = P[j * 3 + 2] - sz;
      float d = dx * dx + dy * dy + dz * dz;
      if (d < dmin[j]) dmin[j] = d;
    }
    __syncthreads();
  }
}

// ---------------------------------------------------------------------------
// Ball query: one block per center; K=64 nearest in-radius points in
// (d2, idx)-lexicographic ascending order == top_k(-d2) semantics.
// ---------------------------------------------------------------------------
__global__ void ball_kernel(const float* __restrict__ pos, const float* __restrict__ centers,
                            int N, int M, float r2,
                            int* __restrict__ nbr, unsigned char* __restrict__ valid) {
  int cidx = blockIdx.x;
  int b = cidx / M;
  const float* P = pos + (size_t)b * N * 3;
  float cx = centers[(size_t)cidx * 3 + 0];
  float cy = centers[(size_t)cidx * 3 + 1];
  float cz = centers[(size_t)cidx * 3 + 2];

  __shared__ float d2s[2048];
  __shared__ float rv[256];
  __shared__ int   ri[256];
  __shared__ float lastd;
  __shared__ int   lasti;
  int t = threadIdx.x;

  for (int j = t; j < N; j += 256) {
    float dx = P[j * 3 + 0] - cx, dy = P[j * 3 + 1] - cy, dz = P[j * 3 + 2] - cz;
    d2s[j] = dx * dx + dy * dy + dz * dz;
  }
  if (t == 0) { lastd = -1.0f; lasti = -1; }
  __syncthreads();

  for (int s = 0; s < 64; ++s) {
    float ld = lastd; int li = lasti;
    float bv = 3.0e38f; int bi = 0x7fffffff;
    for (int j = t; j < N; j += 256) {
      float v = d2s[j];
      if (v <= r2 && (v > ld || (v == ld && j > li))) {
        if (v < bv || (v == bv && j < bi)) { bv = v; bi = j; }
      }
    }
    rv[t] = bv; ri[t] = bi;
    __syncthreads();
    for (int k = 128; k > 0; k >>= 1) {
      if (t < k) {
        float ov = rv[t + k]; int oi = ri[t + k];
        if (ov < rv[t] || (ov == rv[t] && oi < ri[t])) { rv[t] = ov; ri[t] = oi; }
      }
      __syncthreads();
    }
    if (t == 0) {
      size_t o = (size_t)cidx * 64 + s;
      if (rv[0] < 3.0e38f) { nbr[o] = ri[0]; valid[o] = 1; lastd = rv[0]; lasti = ri[0]; }
      else                 { nbr[o] = 0;     valid[o] = 0; }
    }
    __syncthreads();
  }
}

// ---------------------------------------------------------------------------
// Fused SA stage: gather K=64 neighbors into LDS, 3-layer MLP with WMMA
// (B fragment reused across 4 M-tile accumulators), masked max-pool.
// ---------------------------------------------------------------------------
template <int FX, int INPAD, int H1DIM, int H2DIM, int OUTDIM>
__global__ void __launch_bounds__(128)
sa_fused(const _Float16* __restrict__ xfeat, const float* __restrict__ pos,
         const float* __restrict__ centers, const int* __restrict__ nbr,
         const unsigned char* __restrict__ valid,
         const _Float16* __restrict__ Wt0, const float* __restrict__ b0,
         const _Float16* __restrict__ Wt1, const float* __restrict__ b1,
         const _Float16* __restrict__ Wt2, const float* __restrict__ b2,
         _Float16* __restrict__ out, int Npts, int M) {
  int cidx = blockIdx.x;
  int b = cidx / M;

  __shared__ alignas(32) _Float16 Xin[64 * INPAD];
  __shared__ alignas(32) _Float16 Ha[64 * H1DIM];
  __shared__ alignas(32) _Float16 Hb[64 * H2DIM];
  __shared__ int nsh[64];
  __shared__ int vsh[64];

  int t = threadIdx.x;
  float cx = centers[(size_t)cidx * 3 + 0];
  float cy = centers[(size_t)cidx * 3 + 1];
  float cz = centers[(size_t)cidx * 3 + 2];

  size_t base = (size_t)cidx * 64;
  if (t < 64) { nsh[t] = nbr[base + t]; vsh[t] = valid[base + t]; }
  __syncthreads();

  for (int e = t; e < 64 * INPAD; e += 128) {
    int k = e / INPAD;
    int c = e - k * INPAD;
    float val = 0.0f;
    if (vsh[k]) {
      int j = nsh[k];
      if (c < FX) {
        val = (float)xfeat[((size_t)b * Npts + j) * FX + c];
      } else if (c < FX + 3) {
        int d = c - FX;
        float pv = pos[((size_t)b * Npts + j) * 3 + d];
        float cv = (d == 0) ? cx : ((d == 1) ? cy : cz);
        val = pv - cv;
      }
    }
    Xin[e] = (_Float16)val;
  }
  __syncthreads();

  int lane = t & 31;
  int wave = t >> 5;
  int rowL = lane & 15;
  int kh = (lane >> 4) * 16;
  int rhalf = (lane >> 4) * 8;

  constexpr int KT0 = INPAD / 32;
  constexpr int NT1 = H1DIM / 16, KT1 = H1DIM / 32;
  constexpr int NT2 = H2DIM / 16, KT2 = H2DIM / 32;
  constexpr int NT3 = OUTDIM / 16;

  // Layer 1
  for (int nt = wave; nt < NT1; nt += 4) {
    float bb = b0[nt * 16 + rowL];
    v8f acc[4];
#pragma unroll
    for (int mt = 0; mt < 4; ++mt) acc[mt] = (v8f){bb, bb, bb, bb, bb, bb, bb, bb};
#pragma unroll
    for (int kt = 0; kt < KT0; ++kt) {
      v16h w = *(const v16h*)(Wt0 + (size_t)(nt * 16 + rowL) * INPAD + kt * 32 + kh);
#pragma unroll
      for (int mt = 0; mt < 4; ++mt) {
        v16h a = *(const v16h*)(Xin + (mt * 16 + rowL) * INPAD + kt * 32 + kh);
        acc[mt] = WMMA_F16(a, w, acc[mt]);
      }
    }
#pragma unroll
    for (int mt = 0; mt < 4; ++mt)
#pragma unroll
      for (int r = 0; r < 8; ++r) {
        float v = acc[mt][r]; v = v > 0.0f ? v : 0.0f;
        Ha[(mt * 16 + rhalf + r) * H1DIM + nt * 16 + rowL] = (_Float16)v;
      }
  }
  __syncthreads();

  // Layer 2
  for (int nt = wave; nt < NT2; nt += 4) {
    float bb = b1[nt * 16 + rowL];
    v8f acc[4];
#pragma unroll
    for (int mt = 0; mt < 4; ++mt) acc[mt] = (v8f){bb, bb, bb, bb, bb, bb, bb, bb};
#pragma unroll
    for (int kt = 0; kt < KT1; ++kt) {
      v16h w = *(const v16h*)(Wt1 + (size_t)(nt * 16 + rowL) * H1DIM + kt * 32 + kh);
#pragma unroll
      for (int mt = 0; mt < 4; ++mt) {
        v16h a = *(const v16h*)(Ha + (mt * 16 + rowL) * H1DIM + kt * 32 + kh);
        acc[mt] = WMMA_F16(a, w, acc[mt]);
      }
    }
#pragma unroll
    for (int mt = 0; mt < 4; ++mt)
#pragma unroll
      for (int r = 0; r < 8; ++r) {
        float v = acc[mt][r]; v = v > 0.0f ? v : 0.0f;
        Hb[(mt * 16 + rhalf + r) * H2DIM + nt * 16 + rowL] = (_Float16)v;
      }
  }
  __syncthreads();

  // Layer 3 + masked max over 64 neighbor rows
  for (int nt = wave; nt < NT3; nt += 4) {
    float bb = b2[nt * 16 + rowL];
    v8f acc[4];
#pragma unroll
    for (int mt = 0; mt < 4; ++mt) acc[mt] = (v8f){bb, bb, bb, bb, bb, bb, bb, bb};
#pragma unroll
    for (int kt = 0; kt < KT2; ++kt) {
      v16h w = *(const v16h*)(Wt2 + (size_t)(nt * 16 + rowL) * H2DIM + kt * 32 + kh);
#pragma unroll
      for (int mt = 0; mt < 4; ++mt) {
        v16h a = *(const v16h*)(Hb + (mt * 16 + rowL) * H2DIM + kt * 32 + kh);
        acc[mt] = WMMA_F16(a, w, acc[mt]);
      }
    }
    float mx = -__builtin_inff();
#pragma unroll
    for (int mt = 0; mt < 4; ++mt)
#pragma unroll
      for (int r = 0; r < 8; ++r) {
        int row = mt * 16 + rhalf + r;
        if (vsh[row]) { float v = acc[mt][r]; mx = v > mx ? v : mx; }
      }
    float o = __shfl_xor(mx, 16, 32);
    mx = o > mx ? o : mx;
    if (lane < 16) out[(size_t)cidx * OUTDIM + nt * 16 + lane] = (_Float16)mx;
  }
}

// ---------------------------------------------------------------------------
// 64x64-tile WMMA GEMM with TDM-staged A tile in LDS (shared by 4 waves).
// out = act(A[rows,KP] @ Wt[N,KP]^T + bias). grid = (rows/64, Ncols/64).
// ---------------------------------------------------------------------------
__global__ void __launch_bounds__(128)
gemm64(const _Float16* __restrict__ A, const _Float16* __restrict__ Wt,
       const float* __restrict__ bias, _Float16* __restrict__ out,
       int KP, int Ncols, int relu, int kchunk) {
  __shared__ alignas(32) _Float16 Ash[64 * 288];
  int t = threadIdx.x;
  int lane = t & 31, wave = t >> 5;
  int rowBase = blockIdx.x * 64;
  int colBase = blockIdx.y * 64 + wave * 16;
  int rowL = lane & 15, kh = (lane >> 4) * 16;
  float bb = bias[colBase + rowL];
  v8f acc[4];
#pragma unroll
  for (int mt = 0; mt < 4; ++mt) acc[mt] = (v8f){bb, bb, bb, bb, bb, bb, bb, bb};

  int nch = KP / kchunk;
  for (int ch = 0; ch < nch; ++ch) {
#if HAVE_TDM
    if (wave == 0) {
      tdm_load_2d((unsigned)(uintptr_t)(void*)Ash,
                  A + (size_t)rowBase * KP + (size_t)ch * kchunk,
                  kchunk, 64, KP);
      __builtin_amdgcn_s_wait_tensorcnt(0);
    }
#else
    for (int e = t * 8; e < 64 * kchunk; e += 128 * 8) {
      int rr = e / kchunk, cc = e - rr * kchunk;
      *(float4*)(Ash + e) =
          *(const float4*)(A + (size_t)(rowBase + rr) * KP + ch * kchunk + cc);
    }
#endif
    __syncthreads();

    int kts = kchunk / 32;
    for (int kt = 0; kt < kts; ++kt) {
      v16h w = *(const v16h*)(Wt + (size_t)(colBase + rowL) * KP + ch * kchunk + kt * 32 + kh);
#pragma unroll
      for (int mt = 0; mt < 4; ++mt) {
        v16h a = *(const v16h*)(Ash + (mt * 16 + rowL) * kchunk + kt * 32 + kh);
        acc[mt] = WMMA_F16(a, w, acc[mt]);
      }
    }
    __syncthreads();
  }

#pragma unroll
  for (int mt = 0; mt < 4; ++mt)
#pragma unroll
    for (int r = 0; r < 8; ++r) {
      float v = acc[mt][r];
      if (relu) v = v > 0.0f ? v : 0.0f;
      out[(size_t)(rowBase + mt * 16 + (lane >> 4) * 8 + r) * Ncols + colBase + rowL] =
          (_Float16)v;
    }
}

// ---------------------------------------------------------------------------
// Helpers
// ---------------------------------------------------------------------------
__global__ void f32_to_f16(const float* __restrict__ src, _Float16* __restrict__ dst, int n) {
  int i = blockIdx.x * 256 + threadIdx.x;
  if (i < n) dst[i] = (_Float16)src[i];
}

__global__ void wprep(const float* __restrict__ W, _Float16* __restrict__ Wt,
                      int in_dim, int out_dim, int KP) {
  int t = blockIdx.x * 256 + threadIdx.x;
  int total = out_dim * KP;
  if (t < total) {
    int n = t / KP, k = t - n * KP;
    Wt[t] = (k < in_dim) ? (_Float16)W[(size_t)k * out_dim + n] : (_Float16)0.0f;
  }
}

__global__ void sa3pack(const _Float16* __restrict__ x, const float* __restrict__ ctr,
                        _Float16* __restrict__ dst) {
  int t = blockIdx.x * 256 + threadIdx.x;
  int total = 2048 * 288;
  if (t < total) {
    int row = t / 288, c = t - row * 288;
    _Float16 v = (_Float16)0.0f;
    if (c < 256)      v = x[(size_t)row * 256 + c];
    else if (c < 259) v = (_Float16)ctr[(size_t)row * 3 + (c - 256)];
    dst[t] = v;
  }
}

__global__ void maxrows(const _Float16* __restrict__ h, float* __restrict__ g) {
  int t = blockIdx.x * 256 + threadIdx.x;
  int b = t >> 10, c = t & 1023;
  float mx = -__builtin_inff();
  for (int m = 0; m < 256; ++m) {
    float v = (float)h[((size_t)(b * 256 + m)) * 1024 + c];
    mx = v > mx ? v : mx;
  }
  g[t] = mx;
}

__global__ void fc_kernel(const float* __restrict__ g, const float* __restrict__ W,
                          const float* __restrict__ bias, float* __restrict__ out) {
  int t = blockIdx.x * 256 + threadIdx.x;
  int b = t >> 8, j = t & 255;
  float acc = bias[j];
  for (int c = 0; c < 1024; ++c) acc += g[b * 1024 + c] * W[(size_t)c * 256 + j];
  out[t] = acc;
}

// ---------------------------------------------------------------------------
// Host launcher
// ---------------------------------------------------------------------------
extern "C" void kernel_launch(void* const* d_in, const int* in_sizes, int n_in,
                              void* d_out, int out_size, void* d_ws, size_t ws_size,
                              hipStream_t stream) {
  const int B = 8, N = 2048, M1 = 1024, M2 = 256;

  const float* data   = (const float*)d_in[0];
  const float* sa1_w0 = (const float*)d_in[1];  const float* sa1_b0 = (const float*)d_in[2];
  const float* sa1_w1 = (const float*)d_in[3];  const float* sa1_b1 = (const float*)d_in[4];
  const float* sa1_w2 = (const float*)d_in[5];  const float* sa1_b2 = (const float*)d_in[6];
  const float* sa2_w0 = (const float*)d_in[7];  const float* sa2_b0 = (const float*)d_in[8];
  const float* sa2_w1 = (const float*)d_in[9];  const float* sa2_b1 = (const float*)d_in[10];
  const float* sa2_w2 = (const float*)d_in[11]; const float* sa2_b2 = (const float*)d_in[12];
  const float* sa3_w0 = (const float*)d_in[13]; const float* sa3_b0 = (const float*)d_in[14];
  const float* sa3_w1 = (const float*)d_in[15]; const float* sa3_b1 = (const float*)d_in[16];
  const float* sa3_w2 = (const float*)d_in[17]; const float* sa3_b2 = (const float*)d_in[18];
  const float* fc_w   = (const float*)d_in[19]; const float* fc_b   = (const float*)d_in[20];

  char* wsp = (char*)d_ws;
  size_t off = 0;
  auto alloc = [&](size_t bytes) -> void* {
    void* p = wsp + off;
    off += (bytes + 255) & ~(size_t)255;
    return p;
  };

  _Float16* x0h  = (_Float16*)alloc((size_t)B * N * 3 * 2);
  _Float16* Wt10 = (_Float16*)alloc(64  * 32  * 2);
  _Float16* Wt11 = (_Float16*)alloc(64  * 64  * 2);
  _Float16* Wt12 = (_Float16*)alloc(128 * 64  * 2);
  _Float16* Wt20 = (_Float16*)alloc(128 * 160 * 2);
  _Float16* Wt21 = (_Float16*)alloc(128 * 128 * 2);
  _Float16* Wt22 = (_Float16*)alloc(256 * 128 * 2);
  _Float16* Wt30 = (_Float16*)alloc(256 * 288 * 2);
  _Float16* Wt31 = (_Float16*)alloc(512 * 256 * 2);
  _Float16* Wt32 = (_Float16*)alloc((size_t)1024 * 512 * 2);
  int*      idx1 = (int*)alloc((size_t)B * M1 * 4);
  float*    ctr1 = (float*)alloc((size_t)B * M1 * 3 * 4);
  int*      nbr1 = (int*)alloc((size_t)B * M1 * 64 * 4);
  unsigned char* val1 = (unsigned char*)alloc((size_t)B * M1 * 64);
  _Float16* xout1 = (_Float16*)alloc((size_t)B * M1 * 128 * 2);
  int*      idx2 = (int*)alloc((size_t)B * M2 * 4);
  float*    ctr2 = (float*)alloc((size_t)B * M2 * 3 * 4);
  int*      nbr2 = (int*)alloc((size_t)B * M2 * 64 * 4);
  unsigned char* val2 = (unsigned char*)alloc((size_t)B * M2 * 64);
  _Float16* xout2 = (_Float16*)alloc((size_t)B * M2 * 256 * 2);
  _Float16* sa3in = (_Float16*)alloc((size_t)2048 * 288 * 2);
  _Float16* h1    = (_Float16*)alloc((size_t)2048 * 256 * 2);
  _Float16* h2    = (_Float16*)alloc((size_t)2048 * 512 * 2);
  _Float16* h3    = (_Float16*)alloc((size_t)2048 * 1024 * 2);
  float*    g     = (float*)alloc((size_t)B * 1024 * 4);
  (void)ws_size; (void)in_sizes; (void)n_in; (void)out_size;

  f32_to_f16<<<(B * N * 3 + 255) / 256, 256, 0, stream>>>(data, x0h, B * N * 3);
  wprep<<<(64 * 32 + 255) / 256, 256, 0, stream>>>(sa1_w0, Wt10, 6, 64, 32);
  wprep<<<(64 * 64 + 255) / 256, 256, 0, stream>>>(sa1_w1, Wt11, 64, 64, 64);
  wprep<<<(128 * 64 + 255) / 256, 256, 0, stream>>>(sa1_w2, Wt12, 64, 128, 64);
  wprep<<<(128 * 160 + 255) / 256, 256, 0, stream>>>(sa2_w0, Wt20, 131, 128, 160);
  wprep<<<(128 * 128 + 255) / 256, 256, 0, stream>>>(sa2_w1, Wt21, 128, 128, 128);
  wprep<<<(256 * 128 + 255) / 256, 256, 0, stream>>>(sa2_w2, Wt22, 128, 256, 128);
  wprep<<<(256 * 288 + 255) / 256, 256, 0, stream>>>(sa3_w0, Wt30, 259, 256, 288);
  wprep<<<(512 * 256 + 255) / 256, 256, 0, stream>>>(sa3_w1, Wt31, 256, 512, 256);
  wprep<<<(1024 * 512 + 255) / 256, 256, 0, stream>>>(sa3_w2, Wt32, 512, 1024, 512);

  // Stage 1
  fps_kernel<<<B, 256, 0, stream>>>(data, N, M1, idx1, ctr1);
  ball_kernel<<<B * M1, 256, 0, stream>>>(data, ctr1, N, M1, 0.04f, nbr1, val1);
  sa_fused<3, 32, 64, 64, 128><<<B * M1, 128, 0, stream>>>(
      x0h, data, ctr1, nbr1, val1, Wt10, sa1_b0, Wt11, sa1_b1, Wt12, sa1_b2,
      xout1, N, M1);

  // Stage 2
  fps_kernel<<<B, 256, 0, stream>>>(ctr1, M1, M2, idx2, ctr2);
  ball_kernel<<<B * M2, 256, 0, stream>>>(ctr1, ctr2, M1, M2, 0.16f, nbr2, val2);
  sa_fused<128, 160, 128, 128, 256><<<B * M2, 128, 0, stream>>>(
      xout1, ctr1, ctr2, nbr2, val2, Wt20, sa2_b0, Wt21, sa2_b1, Wt22, sa2_b2,
      xout2, M1, M2);

  // Stage 3: global MLP over (xout2 | ctr2), rows = 2048
  sa3pack<<<(2048 * 288 + 255) / 256, 256, 0, stream>>>(xout2, ctr2, sa3in);
  gemm64<<<dim3(32, 4),  128, 0, stream>>>(sa3in, Wt30, sa3_b0, h1, 288, 256, 1, 288);
  gemm64<<<dim3(32, 8),  128, 0, stream>>>(h1,    Wt31, sa3_b1, h2, 256, 512, 1, 256);
  gemm64<<<dim3(32, 16), 128, 0, stream>>>(h2,    Wt32, sa3_b2, h3, 512, 1024, 0, 256);

  // Global max + FC head
  maxrows<<<32, 256, 0, stream>>>(h3, g);
  fc_kernel<<<8, 256, 0, stream>>>(g, fc_w, fc_b, (float*)d_out);
}
